// MergedChunk23Verify_78477642432516
// MI455X (gfx1250) — matrix operand order repaired and checked by reading, not compiled, not run on hardware
//
#include <hip/hip_runtime.h>
#include <hip/hip_bf16.h>
#include <math.h>

// ---------------- types ----------------
typedef __attribute__((ext_vector_type(16))) __bf16   v16bf;
typedef __attribute__((ext_vector_type(8)))  float    v8f;
typedef __attribute__((ext_vector_type(8)))  unsigned v8u;
typedef __attribute__((ext_vector_type(4)))  int      v4i;

#if __has_builtin(__builtin_amdgcn_global_load_async_to_lds_b128) && \
    __has_builtin(__builtin_amdgcn_global_store_async_from_lds_b128) && \
    __has_builtin(__builtin_amdgcn_s_wait_asynccnt)
#define USE_ASYNC_LDS 1
#else
#define USE_ASYNC_LDS 0
#endif

// f32 -> bf16 by truncation. One VALU op per pair via v_perm_b32 (byte pool:
// bytes 0-3 = src1, 4-7 = src0; selector 0x07060302 = {src0.b3,src0.b2,src1.b3,src1.b2}).
__device__ __forceinline__ unsigned short bf16hi(float f) {
    return (unsigned short)(__builtin_bit_cast(unsigned, f) >> 16);
}
__device__ __forceinline__ unsigned pack2(float lo, float hi) {
    return __builtin_amdgcn_perm(__builtin_bit_cast(unsigned, hi),
                                 __builtin_bit_cast(unsigned, lo), 0x07060302u);
}
__device__ __forceinline__ v16bf frag_from(uint4 q0, uint4 q1) {
    v8u r = {q0.x, q0.y, q0.z, q0.w, q1.x, q1.y, q1.z, q1.w};
    return __builtin_bit_cast(v16bf, r);
}

// ---------------- model constants ----------------
#define KLEN 64
#define HDIM 1024
#define NH   8
#define NKV  2
#define HD   128
#define DFF  4096
#define WSL  512
#define CFL  2048

// ---------------- WMMA GEMM: C[64,N] = A[64,K] @ W[K,N], weights streamed exactly once ----------------
// grid = N/64, block = 128 (4 waves); wave w owns columns [16w,16w+16), 4 m-tiles each
__global__ void gemm_wmma_kernel(const float* __restrict__ A, const float* __restrict__ W,
                                 float* __restrict__ C, int N, int K) {
    const int t     = threadIdx.x;
    const int lane  = t & 31;
    const int wave  = t >> 5;
    const int half  = lane >> 4;
    const int lan16 = lane & 15;
    const int n0    = blockIdx.x * 64;

    __shared__ unsigned       Alds[64 * 16];   // [m][k] bf16 pairs: 64 rows x 16 uints (4KB)
    __shared__ unsigned short Wlds[64 * 32];   // transposed [n][k] bf16 (4KB)

    v8f zero = {};
    v8f acc[4];
    #pragma unroll
    for (int i = 0; i < 4; ++i) acc[i] = zero;

    for (int k0 = 0; k0 < K; k0 += 32) {
        // stage A tile (64x32) as packed bf16, row-major
        #pragma unroll
        for (int j = 0; j < 4; ++j) {
            int e = t + j * 128;                 // 0..511 float4 slots
            int m = e >> 3, c4 = e & 7;
            float4 f = *(const float4*)(A + (size_t)m * K + k0 + c4 * 4);
            Alds[m * 16 + c4 * 2]     = pack2(f.x, f.y);
            Alds[m * 16 + c4 * 2 + 1] = pack2(f.z, f.w);
        }
        // stage W tile (32x64) transposed to [n][k] bf16 (wide coalesced b128 loads)
        #pragma unroll
        for (int j = 0; j < 4; ++j) {
            int e = t + j * 128;                 // 0..511
            int kr = e >> 4, c4 = e & 15;
            float4 f = *(const float4*)(W + (size_t)(k0 + kr) * N + n0 + c4 * 4);
            int nb = c4 * 4;
            Wlds[(nb + 0) * 32 + kr] = bf16hi(f.x);
            Wlds[(nb + 1) * 32 + kr] = bf16hi(f.y);
            Wlds[(nb + 2) * 32 + kr] = bf16hi(f.z);
            Wlds[(nb + 3) * 32 + kr] = bf16hi(f.w);
        }
        if (k0 + 32 < K)   // global_prefetch_b8 on next weight tile
            __builtin_prefetch(W + (size_t)(k0 + 32 + (t >> 4)) * N + n0 + (t & 15) * 4, 0, 1);
        __syncthreads();

        // B fragment: one contiguous 32B read (2 x ds_load_b128)
        v16bf bf;
        {
            const uint4* p = (const uint4*)(&Wlds[(size_t)(wave * 16 + lan16) * 32 + half * 16]);
            bf = frag_from(p[0], p[1]);
        }
        #pragma unroll
        for (int mt = 0; mt < 4; ++mt) {
            const uint4* pa = (const uint4*)(&Alds[(size_t)(mt * 16 + lan16) * 16]);
            v16bf af = frag_from(pa[2 * half], pa[2 * half + 1]);
            acc[mt] = __builtin_amdgcn_wmma_f32_16x16x32_bf16(false, af, false, bf,
                                                              (short)0, acc[mt], false, false);
        }
        __syncthreads();
    }
    const int col = n0 + wave * 16 + lan16;
    #pragma unroll
    for (int mt = 0; mt < 4; ++mt) {
        int row = mt * 16 + half * 8;
        #pragma unroll
        for (int r = 0; r < 8; ++r) C[(size_t)(row + r) * N + col] = acc[mt][r];
    }
}

// ---------------- rmsnorm family (H = 1024, one row per block of 256) ----------------
__global__ void add_pl_rmsnorm_kernel(float* __restrict__ h, const float* __restrict__ pl,
                                      const float* __restrict__ w, float* __restrict__ x) {
    const int row = blockIdx.x, t = threadIdx.x;
    __shared__ float red[256];
    float vals[4]; float ss = 0.f;
    #pragma unroll
    for (int i = 0; i < 4; ++i) {
        int c = t + i * 256;
        float v = h[row * HDIM + c] + pl[row * HDIM + c];
        vals[i] = v; h[row * HDIM + c] = v; ss += v * v;
    }
    red[t] = ss; __syncthreads();
    for (int s = 128; s > 0; s >>= 1) { if (t < s) red[t] += red[t + s]; __syncthreads(); }
    float rs = rsqrtf(red[0] / (float)HDIM + 1e-6f);
    #pragma unroll
    for (int i = 0; i < 4; ++i) { int c = t + i * 256; x[row * HDIM + c] = vals[i] * rs * (1.f + w[c]); }
}

__global__ void rmsnorm_kernel(const float* __restrict__ src, const float* __restrict__ w,
                               float* __restrict__ x) {
    const int row = blockIdx.x, t = threadIdx.x;
    __shared__ float red[256];
    float vals[4]; float ss = 0.f;
    #pragma unroll
    for (int i = 0; i < 4; ++i) {
        int c = t + i * 256;
        float v = src[row * HDIM + c];
        vals[i] = v; ss += v * v;
    }
    red[t] = ss; __syncthreads();
    for (int s = 128; s > 0; s >>= 1) { if (t < s) red[t] += red[t + s]; __syncthreads(); }
    float rs = rsqrtf(red[0] / (float)HDIM + 1e-6f);
    #pragma unroll
    for (int i = 0; i < 4; ++i) { int c = t + i * 256; x[row * HDIM + c] = vals[i] * rs * (1.f + w[c]); }
}

__global__ void rmsnorm_residual_kernel(float* __restrict__ h, const float* __restrict__ src,
                                        const float* __restrict__ w) {
    const int row = blockIdx.x, t = threadIdx.x;
    __shared__ float red[256];
    float vals[4]; float ss = 0.f;
    #pragma unroll
    for (int i = 0; i < 4; ++i) {
        int c = t + i * 256;
        float v = src[row * HDIM + c];
        vals[i] = v; ss += v * v;
    }
    red[t] = ss; __syncthreads();
    for (int s = 128; s > 0; s >>= 1) { if (t < s) red[t] += red[t + s]; __syncthreads(); }
    float rs = rsqrtf(red[0] / (float)HDIM + 1e-6f);
    #pragma unroll
    for (int i = 0; i < 4; ++i) { int c = t + i * 256; h[row * HDIM + c] += vals[i] * rs * (1.f + w[c]); }
}

// ---------------- head rmsnorm + RoPE : xin (KLEN, nheads*128) -> out (nheads, KLEN, 128) ----------------
__global__ void rope_norm_kernel(const float* __restrict__ xin, const float* __restrict__ w,
                                 const float* __restrict__ cosb, const float* __restrict__ sinb,
                                 float* __restrict__ out, int nheads) {
    const int hh  = blockIdx.x >> 6;
    const int pos = blockIdx.x & 63;
    const int d   = threadIdx.x;   // 128
    __shared__ float red[128];
    __shared__ float sn[128];
    float v = xin[pos * (nheads * HD) + hh * HD + d];
    red[d] = v * v; __syncthreads();
    for (int s = 64; s > 0; s >>= 1) { if (d < s) red[d] += red[d + s]; __syncthreads(); }
    float rs = rsqrtf(red[0] / (float)HD + 1e-6f);
    float xn = v * rs * (1.f + w[d]);
    sn[d] = xn; __syncthreads();
    float rot = (d < 64) ? -sn[d + 64] : sn[d - 64];
    out[(size_t)(hh * KLEN + pos) * HD + d] = xn * cosb[pos * HD + d] + rot * sinb[pos * HD + d];
}

// V reshape: xv (KLEN, NKV*128) -> out (NKV, KLEN, 128)
__global__ void reshape_v_kernel(const float* __restrict__ xv, float* __restrict__ out) {
    int i = blockIdx.x * blockDim.x + threadIdx.x;
    if (i >= NKV * KLEN * HD) return;
    int d = i & 127; int r = i >> 7; int pos = r & 63; int kvh = r >> 6;
    out[i] = xv[pos * (NKV * HD) + kvh * HD + d];
}

// gated cache output: dst = src * gate[pos], src layout (NKV, KLEN, HD)
__global__ void gate_scale_kernel(const float* __restrict__ src, const float* __restrict__ gate,
                                  float* __restrict__ dst) {
    int i = blockIdx.x * blockDim.x + threadIdx.x;
    if (i >= NKV * KLEN * HD) return;
    int pos = (i >> 7) & 63;
    dst[i] = src[i] * gate[pos];
}

// concat along positions: dst (NKV, Lc+64, HD) = [cache ; knew] — async global->LDS->global path
__global__ void concat_kv_kernel(float* __restrict__ dst, const float* __restrict__ cache,
                                 const float* __restrict__ knew, int Lc) {
    const int nvec = NKV * (Lc + KLEN) * (HD / 4);   // float4 count
#if USE_ASYNC_LDS
    __shared__ float4 buf[256];
    for (int base = blockIdx.x * 256; base < nvec; base += gridDim.x * 256) {
        int i = base + threadIdx.x;
        if (i < nvec) {
            int r = i >> 5, c4 = i & 31;              // 32 float4 per 128-elem row
            int pos = r % (Lc + KLEN), kvh = r / (Lc + KLEN);
            const float* src = (pos < Lc)
                ? cache + ((size_t)kvh * Lc + pos) * HD + c4 * 4
                : knew  + ((size_t)kvh * KLEN + (pos - Lc)) * HD + c4 * 4;
            __builtin_amdgcn_global_load_async_to_lds_b128(
                (v4i*)src, (v4i*)&buf[threadIdx.x], 0, 0);
        }
        __builtin_amdgcn_s_wait_asynccnt(0);
        if (i < nvec) {
            __builtin_amdgcn_global_store_async_from_lds_b128(
                (v4i*)(dst + (size_t)i * 4), (v4i*)&buf[threadIdx.x], 0, 0);
        }
        __builtin_amdgcn_s_wait_asynccnt(0);
    }
#else
    for (int i = blockIdx.x * blockDim.x + threadIdx.x; i < nvec; i += gridDim.x * blockDim.x) {
        int r = i >> 5, c4 = i & 31;
        int pos = r % (Lc + KLEN), kvh = r / (Lc + KLEN);
        const float* src = (pos < Lc)
            ? cache + ((size_t)kvh * Lc + pos) * HD + c4 * 4
            : knew  + ((size_t)kvh * KLEN + (pos - Lc)) * HD + c4 * 4;
        *(float4*)(dst + (size_t)i * 4) = *(const float4*)src;
    }
#endif
}

// gelu(tanh approx)(g) * u
__global__ void gelu_mul_kernel(const float* __restrict__ g, const float* __restrict__ u,
                                float* __restrict__ t, int n) {
    int i = blockIdx.x * blockDim.x + threadIdx.x;
    if (i >= n) return;
    float x = g[i];
    float c = 0.7978845608028654f * (x + 0.044715f * x * x * x);
    float gl = 0.5f * x * (1.f + tanhf(c));
    t[i] = gl * u[i];
}

// ---------------- flash attention, WMMA bf16, wide staged LDS ----------------
// grid = NH, block = 128 (wave w handles q rows [16w,16w+16))
__global__ void attn_kernel(const float* __restrict__ Q,
                            const float* __restrict__ Kc, const float* __restrict__ Vc, int Lc,
                            const float* __restrict__ Kn, const float* __restrict__ Vn,
                            const float* __restrict__ mask, int Lk,
                            float* __restrict__ O, float scale) {
    const int h     = blockIdx.x;
    const int kvh   = h >> 2;
    const int t     = threadIdx.x;
    const int lane  = t & 31;
    const int wave  = t >> 5;
    const int half  = lane >> 4;
    const int lan16 = lane & 15;

    __shared__ unsigned       Klds[32 * 64];       // [key][d] bf16 pairs (8KB)
    __shared__ unsigned short Vlds[128 * 32];      // transposed [d][key] bf16 (8KB)
    __shared__ unsigned short Sb[4][16 * 32];      // per-wave P tile bf16 (4KB)

    // Q A-fragments (d0 = 0,32,64,96) via float4 loads + packed convert
    v16bf qf[4];
    {
        const float* qrow = Q + (size_t)(h * KLEN + wave * 16 + lan16) * HD;
        #pragma unroll
        for (int tt = 0; tt < 4; ++tt) {
            int kb = tt * 32 + half * 16;
            float4 f0 = *(const float4*)(qrow + kb);
            float4 f1 = *(const float4*)(qrow + kb + 4);
            float4 f2 = *(const float4*)(qrow + kb + 8);
            float4 f3 = *(const float4*)(qrow + kb + 12);
            v8u r = {pack2(f0.x, f0.y), pack2(f0.z, f0.w), pack2(f1.x, f1.y), pack2(f1.z, f1.w),
                     pack2(f2.x, f2.y), pack2(f2.z, f2.w), pack2(f3.x, f3.y), pack2(f3.z, f3.w)};
            qf[tt] = __builtin_bit_cast(v16bf, r);
        }
    }

    v8f zero = {};
    v8f oacc[8];
    #pragma unroll
    for (int i = 0; i < 8; ++i) oacc[i] = zero;
    float rmax[8], rsum[8];
    #pragma unroll
    for (int r = 0; r < 8; ++r) { rmax[r] = -1e30f; rsum[r] = 0.f; }

    for (int kk = 0; kk < Lk; kk += 32) {
        // stage 32-key K/V chunk: float4 loads, packed bf16 stores (V transposed)
        #pragma unroll
        for (int j = 0; j < 8; ++j) {
            int e = t + j * 128;             // 0..1023 float4 slots
            int keyl = e >> 5, c4 = e & 31;
            int key = kk + keyl;
            const float *ks, *vs;
            if (key < Lc) {
                ks = Kc + ((size_t)kvh * Lc + key) * HD + c4 * 4;
                vs = Vc + ((size_t)kvh * Lc + key) * HD + c4 * 4;
            } else {
                ks = Kn + ((size_t)kvh * KLEN + (key - Lc)) * HD + c4 * 4;
                vs = Vn + ((size_t)kvh * KLEN + (key - Lc)) * HD + c4 * 4;
            }
            float4 f = *(const float4*)ks;
            Klds[keyl * 64 + c4 * 2]     = pack2(f.x, f.y);
            Klds[keyl * 64 + c4 * 2 + 1] = pack2(f.z, f.w);
            float4 g = *(const float4*)vs;
            int d0 = c4 * 4;
            Vlds[(d0 + 0) * 32 + keyl] = bf16hi(g.x);
            Vlds[(d0 + 1) * 32 + keyl] = bf16hi(g.y);
            Vlds[(d0 + 2) * 32 + keyl] = bf16hi(g.z);
            Vlds[(d0 + 3) * 32 + keyl] = bf16hi(g.w);
        }
        __syncthreads();

        // S = Q @ K^T : contiguous 32B ds reads per fragment
        v8f s0 = zero, s1 = zero;
        #pragma unroll
        for (int tt = 0; tt < 4; ++tt) {
            int du = tt * 16 + half * 8;      // uint offset = d/2
            const uint4* p0 = (const uint4*)(&Klds[(size_t)lan16 * 64 + du]);
            const uint4* p1 = (const uint4*)(&Klds[(size_t)(16 + lan16) * 64 + du]);
            v16bf b0 = frag_from(p0[0], p0[1]);
            v16bf b1 = frag_from(p1[0], p1[1]);
            s0 = __builtin_amdgcn_wmma_f32_16x16x32_bf16(false, qf[tt], false, b0, (short)0, s0, false, false);
            s1 = __builtin_amdgcn_wmma_f32_16x16x32_bf16(false, qf[tt], false, b1, (short)0, s1, false, false);
        }

        // online softmax
        float alpha[8];
        #pragma unroll
        for (int r = 0; r < 8; ++r) {
            int row = wave * 16 + half * 8 + r;
            float a0 = s0[r] * scale + mask[(size_t)row * Lk + kk + lan16];
            float a1 = s1[r] * scale + mask[(size_t)row * Lk + kk + 16 + lan16];
            float mx = fmaxf(a0, a1);
            #pragma unroll
            for (int off = 1; off < 16; off <<= 1) mx = fmaxf(mx, __shfl_xor(mx, off, 32));
            float nm = fmaxf(rmax[r], mx);
            alpha[r] = __expf(rmax[r] - nm);
            float p0 = __expf(a0 - nm);
            float p1 = __expf(a1 - nm);
            float sr = p0 + p1;
            #pragma unroll
            for (int off = 1; off < 16; off <<= 1) sr += __shfl_xor(sr, off, 32);
            rsum[r] = rsum[r] * alpha[r] + sr;
            rmax[r] = nm;
            int rowloc = half * 8 + r;
            Sb[wave][rowloc * 32 + lan16]      = bf16hi(p0);
            Sb[wave][rowloc * 32 + 16 + lan16] = bf16hi(p1);
        }
        #pragma unroll
        for (int tt = 0; tt < 8; ++tt) {
            #pragma unroll
            for (int r = 0; r < 8; ++r) oacc[tt][r] *= alpha[r];
        }
        __syncthreads();

        // P in A-fragment layout straight from bf16 LDS tile
        v16bf pf;
        {
            const uint4* ps = (const uint4*)(&Sb[wave][(size_t)lan16 * 32 + half * 8]);
            pf = frag_from(ps[0], ps[2]);     // +0B: k..k+7, +32B: k+16..k+23
        }
        // O += P @ V (transposed V tile: two 16B reads, 32B apart)
        #pragma unroll
        for (int tt = 0; tt < 8; ++tt) {
            int n = tt * 16 + lan16;
            const uint4* pv = (const uint4*)(&Vlds[(size_t)n * 32 + half * 8]);
            v16bf bv = frag_from(pv[0], pv[2]);
            oacc[tt] = __builtin_amdgcn_wmma_f32_16x16x32_bf16(false, pf, false, bv, (short)0, oacc[tt], false, false);
        }
        __syncthreads();
    }

    // epilogue: normalize, write O (KLEN, NH*HD) in [pos][h*128+d] layout
    #pragma unroll
    for (int tt = 0; tt < 8; ++tt) {
        #pragma unroll
        for (int r = 0; r < 8; ++r) {
            int row = wave * 16 + half * 8 + r;
            O[(size_t)row * (NH * HD) + h * HD + tt * 16 + lan16] = oacc[tt][r] / rsum[r];
        }
    }
}

// ---------------- host orchestration ----------------
extern "C" void kernel_launch(void* const* d_in, const int* in_sizes, int n_in,
                              void* d_out, int out_size, void* d_ws, size_t ws_size,
                              hipStream_t stream) {
    const float* hidden    = (const float*)d_in[0];
    const float* mask_f    = (const float*)d_in[1];   // (64, 2112)
    const float* mask_s    = (const float*)d_in[2];   // (64, 576)
    const float* gate      = (const float*)d_in[3];   // (64,)
    const float* plc       = (const float*)d_in[4];   // (25, 64, 1024)
    const float* cos_s     = (const float*)d_in[5];
    const float* sin_s     = (const float*)d_in[6];
    const float* cos_f     = (const float*)d_in[7];
    const float* sin_f     = (const float*)d_in[8];
    const float* Ks_in     = (const float*)d_in[9];   // (5, 2, 512, 128)
    const float* Vs_in     = (const float*)d_in[10];
    const float* Kf_in     = (const float*)d_in[11];  // (2, 2, 2048, 128)
    const float* Vf_in     = (const float*)d_in[12];
    const float* Wq        = (const float*)d_in[13];
    const float* Wk        = (const float*)d_in[14];
    const float* Wv        = (const float*)d_in[15];
    const float* Wo        = (const float*)d_in[16];
    const float* Wg        = (const float*)d_in[17];
    const float* Wu        = (const float*)d_in[18];
    const float* Wd        = (const float*)d_in[19];
    const float* ln_in     = (const float*)d_in[20];
    const float* ln_pa     = (const float*)d_in[21];
    const float* ln_pf     = (const float*)d_in[22];
    const float* ln_pff    = (const float*)d_in[23];
    const float* q_norm    = (const float*)d_in[24];
    const float* k_norm    = (const float*)d_in[25];

    float* out = (float*)d_out;
    const size_t O_SK  = 65536;
    const size_t O_SV  = O_SK  + (size_t)5 * NKV * KLEN * HD;
    const size_t O_FK  = O_SV  + (size_t)5 * NKV * KLEN * HD;
    const size_t O_FV  = O_FK  + (size_t)2 * NKV * KLEN * HD;
    const size_t O_K13 = O_FV  + (size_t)2 * NKV * KLEN * HD;
    const size_t O_V13 = O_K13 + (size_t)NKV * (WSL + KLEN) * HD;
    const size_t O_K14 = O_V13 + (size_t)NKV * (WSL + KLEN) * HD;
    const size_t O_V14 = O_K14 + (size_t)NKV * (CFL + KLEN) * HD;

    float* ws = (float*)d_ws;
    float* h  = ws;
    float* x  = h  + 65536;
    float* xq = x  + 65536;
    float* qb = xq + 65536;
    float* xk = qb + 65536;
    float* xv = xk + 16384;
    float* kb = xv + 16384;
    float* vb = kb + 16384;
    float* ao = vb + 16384;
    float* ob = ao + 65536;
    float* gb = ob + 65536;
    float* ub = gb + 262144;
    float* tb = ub + 262144;
    float* mb = tb + 262144;

    (void)hipMemcpyAsync(h, hidden, (size_t)KLEN * HDIM * sizeof(float),
                         hipMemcpyDeviceToDevice, stream);

    const float scale = 1.0f / sqrtf((float)HD);
    const int smap[7] = {0, -1, 1, 2, 3, 4, -1};
    const int fmap[7] = {-1, 0, -1, -1, -1, -1, 1};

    for (int layer = 8; layer < 25; ++layer) {
        const int li = layer - 8;
        const bool sliding = (layer % 5) != 4;
        const float* cosb = sliding ? cos_s : cos_f;
        const float* sinb = sliding ? sin_s : sin_f;
        const float* mask = sliding ? mask_s : mask_f;
        const int Lk = sliding ? (WSL + KLEN) : (CFL + KLEN);

        add_pl_rmsnorm_kernel<<<KLEN, 256, 0, stream>>>(h, plc + (size_t)layer * KLEN * HDIM,
                                                        ln_in + (size_t)li * HDIM, x);
        gemm_wmma_kernel<<<16, 128, 0, stream>>>(x, Wq + (size_t)li * HDIM * (NH * HD),
                                                 xq, NH * HD, HDIM);
        rope_norm_kernel<<<NH * KLEN, HD, 0, stream>>>(xq, q_norm + (size_t)li * HD, cosb, sinb, qb, NH);

        const float *kc, *vc, *kn, *vn; int Lc;
        if (layer < 15) {
            gemm_wmma_kernel<<<4, 128, 0, stream>>>(x, Wk + (size_t)li * HDIM * (NKV * HD),
                                                    xk, NKV * HD, HDIM);
            gemm_wmma_kernel<<<4, 128, 0, stream>>>(x, Wv + (size_t)li * HDIM * (NKV * HD),
                                                    xv, NKV * HD, HDIM);
            rope_norm_kernel<<<NKV * KLEN, HD, 0, stream>>>(xk, k_norm + (size_t)li * HD, cosb, sinb, kb, NKV);
            reshape_v_kernel<<<(NKV * KLEN * HD + 255) / 256, 256, 0, stream>>>(xv, vb);

            if (sliding) {
                int slot = smap[li];
                gate_scale_kernel<<<(NKV * KLEN * HD + 255) / 256, 256, 0, stream>>>(
                    kb, gate, out + O_SK + (size_t)slot * NKV * KLEN * HD);
                gate_scale_kernel<<<(NKV * KLEN * HD + 255) / 256, 256, 0, stream>>>(
                    vb, gate, out + O_SV + (size_t)slot * NKV * KLEN * HD);
            } else {
                int slot = fmap[li];
                gate_scale_kernel<<<(NKV * KLEN * HD + 255) / 256, 256, 0, stream>>>(
                    kb, gate, out + O_FK + (size_t)slot * NKV * KLEN * HD);
                gate_scale_kernel<<<(NKV * KLEN * HD + 255) / 256, 256, 0, stream>>>(
                    vb, gate, out + O_FV + (size_t)slot * NKV * KLEN * HD);
            }

            if (layer == 13) {
                concat_kv_kernel<<<512, 256, 0, stream>>>(out + O_K13,
                    Ks_in + (size_t)smap[li] * NKV * WSL * HD, kb, WSL);
                concat_kv_kernel<<<512, 256, 0, stream>>>(out + O_V13,
                    Vs_in + (size_t)smap[li] * NKV * WSL * HD, vb, WSL);
                kc = out + O_K13; vc = out + O_V13; Lc = WSL + KLEN; kn = kb; vn = vb;
            } else if (layer == 14) {
                concat_kv_kernel<<<528, 256, 0, stream>>>(out + O_K14,
                    Kf_in + (size_t)fmap[li] * NKV * CFL * HD, kb, CFL);
                concat_kv_kernel<<<528, 256, 0, stream>>>(out + O_V14,
                    Vf_in + (size_t)fmap[li] * NKV * CFL * HD, vb, CFL);
                kc = out + O_K14; vc = out + O_V14; Lc = CFL + KLEN; kn = kb; vn = vb;
            } else if (sliding) {
                int slot = smap[li];
                kc = Ks_in + (size_t)slot * NKV * WSL * HD;
                vc = Vs_in + (size_t)slot * NKV * WSL * HD;
                Lc = WSL; kn = kb; vn = vb;
            } else {
                int slot = fmap[li];
                kc = Kf_in + (size_t)slot * NKV * CFL * HD;
                vc = Vf_in + (size_t)slot * NKV * CFL * HD;
                Lc = CFL; kn = kb; vn = vb;
            }
        } else {
            if (sliding) { kc = out + O_K13; vc = out + O_V13; Lc = WSL + KLEN; }
            else         { kc = out + O_K14; vc = out + O_V14; Lc = CFL + KLEN; }
            kn = kb; vn = vb;
        }

        attn_kernel<<<NH, 128, 0, stream>>>(qb, kc, vc, Lc, kn, vn, mask, Lk, ao, scale);

        gemm_wmma_kernel<<<16, 128, 0, stream>>>(ao, Wo + (size_t)li * (NH * HD) * HDIM,
                                                 ob, HDIM, NH * HD);
        rmsnorm_residual_kernel<<<KLEN, 256, 0, stream>>>(h, ob, ln_pa + (size_t)li * HDIM);

        rmsnorm_kernel<<<KLEN, 256, 0, stream>>>(h, ln_pf + (size_t)li * HDIM, x);
        gemm_wmma_kernel<<<64, 128, 0, stream>>>(x, Wg + (size_t)li * HDIM * DFF, gb, DFF, HDIM);
        gemm_wmma_kernel<<<64, 128, 0, stream>>>(x, Wu + (size_t)li * HDIM * DFF, ub, DFF, HDIM);
        gelu_mul_kernel<<<(KLEN * DFF + 255) / 256, 256, 0, stream>>>(gb, ub, tb, KLEN * DFF);
        gemm_wmma_kernel<<<16, 128, 0, stream>>>(tb, Wd + (size_t)li * DFF * HDIM, mb, HDIM, DFF);
        rmsnorm_residual_kernel<<<KLEN, 256, 0, stream>>>(h, mb, ln_pff + (size_t)li * HDIM);
    }

    (void)hipMemcpyAsync(out, h, (size_t)KLEN * HDIM * sizeof(float),
                         hipMemcpyDeviceToDevice, stream);
}
